// Temporal_SelfAttention_56212531970123
// MI455X (gfx1250) — matrix operand order repaired and checked by reading, not compiled, not run on hardware
//
#include <hip/hip_runtime.h>

typedef __attribute__((ext_vector_type(16))) __bf16 v16bf;
typedef __attribute__((ext_vector_type(8)))  float  v8f;
typedef __attribute__((ext_vector_type(4)))  unsigned int u32x4;
typedef __attribute__((ext_vector_type(8)))  int i32x8;
typedef __attribute__((ext_vector_type(4)))  int i32x4;

#define B_  2
#define H_  8
#define L_  2048
#define D_  512
#define DH_ 64

#if defined(__has_builtin)
#if __has_builtin(__builtin_amdgcn_tensor_load_to_lds)
#define HAVE_TDM 1
#endif
#if __has_builtin(__builtin_amdgcn_s_wait_tensorcnt)
#define HAVE_TENSORCNT 1
#endif
#endif

__device__ __forceinline__ v8f wmma_bf16(v16bf a, v16bf b, v8f c) {
  return __builtin_amdgcn_wmma_f32_16x16x32_bf16(false, a, false, b, (short)0, c,
                                                 false, false);
}

// A-fragment (16x32 bf16), row-major source (m,k) at src[m*ld+k].
__device__ __forceinline__ v16bf load_a_bf16(const __bf16* src, int ld, int lane) {
  const int m  = lane & 15;
  const int kb = (lane & 16) ? 8 : 0;
  const __bf16* r = src + (size_t)m * ld;
  v16bf a;
#pragma unroll
  for (int i = 0; i < 8; ++i) a[i] = r[kb + i];
#pragma unroll
  for (int i = 0; i < 8; ++i) a[8 + i] = r[16 + kb + i];
  return a;
}

__device__ __forceinline__ v16bf load_a_f32(const float* src, size_t ld, int lane) {
  const int m  = lane & 15;
  const int kb = (lane & 16) ? 8 : 0;
  const float* r = src + (size_t)m * ld;
  v16bf a;
#pragma unroll
  for (int i = 0; i < 8; ++i) a[i] = (__bf16)r[kb + i];
#pragma unroll
  for (int i = 0; i < 8; ++i) a[8 + i] = (__bf16)r[16 + kb + i];
  return a;
}

// B-fragment (32x16 bf16), N-major source with K contiguous: (k,n) at src[n*ld+k].
__device__ __forceinline__ v16bf load_b_bf16(const __bf16* src, size_t ld, int lane) {
  const int n  = lane & 15;
  const int kb = (lane & 16) ? 8 : 0;
  const __bf16* r = src + (size_t)n * ld + kb;
  v16bf b;
#pragma unroll
  for (int i = 0; i < 8; ++i) b[i] = r[i];
#pragma unroll
  for (int i = 0; i < 8; ++i) b[8 + i] = r[16 + i];
  return b;
}

// TDM: async DMA of a 2D bf16 tile (tile_d0 x tile_d1, row stride stride0 elems)
// from global memory into LDS at byte offset lds_off. Issued by one wave.
__device__ __forceinline__ void tdm_load_tile_2d_bf16(unsigned lds_off, const void* gaddr,
                                                      unsigned tensor_d0, unsigned tensor_d1,
                                                      unsigned tile_d0, unsigned tile_d1,
                                                      unsigned stride0) {
#ifdef HAVE_TDM
  unsigned long long ga = (unsigned long long)gaddr;
  u32x4 g0;
  g0[0] = 1u;                                               // count=1 user descriptor
  g0[1] = lds_off;                                          // lds_addr (bytes)
  g0[2] = (unsigned)(ga & 0xFFFFFFFFu);                     // global_addr[31:0]
  g0[3] = (unsigned)((ga >> 32) & 0x1FFFFFFu) | (2u << 30); // addr[56:32] | type=2
  i32x8 g1;
  g1[0] = (int)(1u << 16);                                  // data_size=1 -> 2 bytes
  g1[1] = (int)((tensor_d0 & 0xFFFFu) << 16);               // tensor_dim0[15:0]
  g1[2] = (int)(((tensor_d0 >> 16) & 0xFFFFu) |
                ((tensor_d1 & 0xFFFFu) << 16));             // dim0 hi | dim1 lo
  g1[3] = (int)(((tensor_d1 >> 16) & 0xFFFFu) |
                ((tile_d0 & 0xFFFFu) << 16));               // dim1 hi | tile_dim0
  g1[4] = (int)(tile_d1 & 0xFFFFu);                         // tile_dim1 (tile_dim2=0)
  g1[5] = (int)stride0;                                     // tensor_dim0_stride[31:0]
  g1[6] = 0;
  g1[7] = 0;
  i32x4 z4 = {0, 0, 0, 0};
#if __clang_major__ >= 23
  i32x8 z8 = {0, 0, 0, 0, 0, 0, 0, 0};
  __builtin_amdgcn_tensor_load_to_lds(g0, g1, z4, z4, z8, 0);
#else
  __builtin_amdgcn_tensor_load_to_lds(g0, g1, z4, z4, 0);
#endif
#ifdef HAVE_TENSORCNT
  __builtin_amdgcn_s_wait_tensorcnt(0);
#else
  asm volatile("s_wait_tensorcnt 0x0" ::: "memory");
#endif
#endif
}

__device__ __forceinline__ float half_reduce_max(float v) {
  v = fmaxf(v, __shfl_xor(v, 1, 32));
  v = fmaxf(v, __shfl_xor(v, 2, 32));
  v = fmaxf(v, __shfl_xor(v, 4, 32));
  v = fmaxf(v, __shfl_xor(v, 8, 32));   // masks<16: never mixes the two row-halves
  return v;
}
__device__ __forceinline__ float half_reduce_sum(float v) {
  v += __shfl_xor(v, 1, 32);
  v += __shfl_xor(v, 2, 32);
  v += __shfl_xor(v, 4, 32);
  v += __shfl_xor(v, 8, 32);
  return v;
}

// ---------------- Kernel 1: convert/transpose weights + x to bf16 --------------
__global__ void prep_kernel(const float* __restrict__ x,  const float* __restrict__ Wq,
                            const float* __restrict__ Wk, const float* __restrict__ Wv,
                            const float* __restrict__ Wo, const float* __restrict__ We,
                            __bf16* xb, __bf16* wqT, __bf16* wkT, __bf16* wvT,
                            __bf16* woT, __bf16* weT) {
  const int NX = B_ * L_ * D_;
  const int NW = D_ * D_;
  const int NE = L_ * DH_;
  const int total = NX + 4 * NW + NE;
  for (int idx = blockIdx.x * blockDim.x + threadIdx.x; idx < total;
       idx += gridDim.x * blockDim.x) {
    if (idx < NX) { xb[idx] = (__bf16)x[idx]; continue; }
    int t = idx - NX;
    if (t < 4 * NW) {
      int w = t / NW, r = t - w * NW;
      int o = r >> 9, i = r & 511;
      const float* W = (w == 0) ? Wq : (w == 1) ? Wk : (w == 2) ? Wv : Wo;
      __bf16* WT = (w == 0) ? wqT : (w == 1) ? wkT : (w == 2) ? wvT : woT;
      WT[r] = (__bf16)W[i * D_ + o];
      continue;
    }
    t -= 4 * NW;
    int f = t >> 6, d = t & 63;
    weT[t] = (__bf16)We[d * L_ + f];
  }
}

// ---------------- Kernel 2: QKV projection (WMMA) ------------------------------
__global__ void qkv_kernel(const __bf16* __restrict__ xb, const __bf16* __restrict__ wqT,
                           const __bf16* __restrict__ wkT, const __bf16* __restrict__ wvT,
                           __bf16* q, __bf16* k, __bf16* vT) {
  const int lane = threadIdx.x & 31;
  const int wave = threadIdx.x >> 5;
  const int nw   = blockIdx.y * 8 + wave;   // 0..95
  const int wsel = nw >> 5;
  const int dout0 = (nw & 31) * 16;
  const int row0  = blockIdx.x * 16;

  const __bf16* WT = (wsel == 0) ? wqT : (wsel == 1) ? wkT : wvT;
  v8f acc = {};
  for (int kk = 0; kk < 16; ++kk) {
    v16bf a = load_a_bf16(xb + (size_t)row0 * D_ + kk * 32, D_, lane);
    v16bf b = load_b_bf16(WT + (size_t)dout0 * D_ + kk * 32, D_, lane);
    acc = wmma_bf16(a, b, acc);
  }
  const int mh   = (lane & 16) ? 8 : 0;
  const int dout = dout0 + (lane & 15);
  const int h = dout >> 6, dd = dout & 63;
#pragma unroll
  for (int r = 0; r < 8; ++r) {
    int row = row0 + r + mh;
    int bb = row >> 11, l = row & (L_ - 1);
    __bf16 val = (__bf16)acc[r];
    size_t bh = (size_t)(bb * H_ + h);
    if (wsel == 0)      q [(bh * L_ + l) * DH_ + dd] = val;
    else if (wsel == 1) k [(bh * L_ + l) * DH_ + dd] = val;
    else                vT[(bh * DH_ + dd) * (size_t)L_ + l] = val;
  }
}

// ------- Kernel 3: fused e-tile + scores*s_rel*scale + row softmax -> A --------
// grid (128 l-tiles, 16 bh), block 256 (8 waves). s_rel[l,s] = e[l, 2047-|s-l|].
// Each wave keeps its 16 score tiles (128 VGPRs) in registers through softmax,
// so A is written exactly once (post-softmax): halves A-matrix HBM traffic.
__global__ void scores_softmax_kernel(const __bf16* __restrict__ q,
                                      const __bf16* __restrict__ k,
                                      const __bf16* __restrict__ weT,
                                      float* __restrict__ A) {
  __shared__ __align__(16) __bf16 qtile[16][DH_];  // 2 KB (TDM destination)
  __shared__ __bf16 e_lds[16][L_];                 // 64 KB
  __shared__ float red[16][8];
  const int lane = threadIdx.x & 31;
  const int wave = threadIdx.x >> 5;
  const int bh = blockIdx.y;
  const int l0 = blockIdx.x * 16;
  const int mh = (lane & 16) ? 8 : 0;
  const int nlo = lane & 15;

  const __bf16* qh = q + (size_t)bh * L_ * DH_;
  const __bf16* kh = k + (size_t)bh * L_ * DH_;

  // Stage the 16x64 q-tile into LDS with the Tensor Data Mover (wave 0 only).
  if (wave == 0) {
#ifdef HAVE_TDM
    unsigned loff = (unsigned)(size_t)(void*)&qtile[0][0];
    tdm_load_tile_2d_bf16(loff, qh + (size_t)l0 * DH_,
                          /*tensor_d0=*/DH_, /*tensor_d1=*/(unsigned)(L_ - l0),
                          /*tile_d0=*/DH_, /*tile_d1=*/16, /*stride0=*/DH_);
#else
    for (int i = lane; i < 16 * DH_; i += 32)
      qtile[i >> 6][i & 63] = qh[(size_t)(l0 + (i >> 6)) * DH_ + (i & 63)];
#endif
  }
  __syncthreads();

  v16bf a0 = load_a_bf16(&qtile[0][0],      DH_, lane);   // K = d 0..31
  v16bf a1 = load_a_bf16(&qtile[0][0] + 32, DH_, lane);   // K = d 32..63

  // phase 1: e_tile[16][2048] = q_tile @ We
  for (int t = 0; t < 16; ++t) {
    int f0 = (wave * 16 + t) * 16;
    v16bf b0 = load_b_bf16(weT + (size_t)f0 * DH_,      DH_, lane);
    v16bf b1 = load_b_bf16(weT + (size_t)f0 * DH_ + 32, DH_, lane);
    v8f acc = {};
    acc = wmma_bf16(a0, b0, acc);
    acc = wmma_bf16(a1, b1, acc);
#pragma unroll
    for (int r = 0; r < 8; ++r)
      e_lds[r + mh][f0 + nlo] = (__bf16)acc[r];
  }
  __syncthreads();

  // phase 2: logits tiles in registers: scale * (q.kT) * s_rel
  const float scale = 0.125f;               // 1/sqrt(64)
  v8f sc[16];
#pragma unroll
  for (int t = 0; t < 16; ++t) {
    int s0 = (wave * 16 + t) * 16;
    v16bf b0 = load_b_bf16(kh + (size_t)s0 * DH_,      DH_, lane);
    v16bf b1 = load_b_bf16(kh + (size_t)s0 * DH_ + 32, DH_, lane);
    v8f acc = {};
    acc = wmma_bf16(a0, b0, acc);
    acc = wmma_bf16(a1, b1, acc);
    int s = s0 + nlo;
#pragma unroll
    for (int r = 0; r < 8; ++r) {
      int m = r + mh;
      int l = l0 + m;
      int rel = 2047 - (s > l ? s - l : l - s);
      acc[r] = scale * acc[r] * (float)e_lds[m][rel];
    }
    sc[t] = acc;
  }

  // phase 3: row softmax across the whole s-dimension (in-register + LDS)
  float rowmax[8];
#pragma unroll
  for (int r = 0; r < 8; ++r) {
    float m_ = sc[0][r];
#pragma unroll
    for (int t = 1; t < 16; ++t) m_ = fmaxf(m_, sc[t][r]);
    m_ = half_reduce_max(m_);
    rowmax[r] = m_;
  }
  if (nlo == 0) {
#pragma unroll
    for (int r = 0; r < 8; ++r) red[r + mh][wave] = rowmax[r];
  }
  __syncthreads();
#pragma unroll
  for (int r = 0; r < 8; ++r) {
    float m_ = red[r + mh][0];
#pragma unroll
    for (int w = 1; w < 8; ++w) m_ = fmaxf(m_, red[r + mh][w]);
    rowmax[r] = m_;
  }
  __syncthreads();                           // before reusing red for sums

  float rowsum[8];
#pragma unroll
  for (int r = 0; r < 8; ++r) rowsum[r] = 0.f;
#pragma unroll
  for (int t = 0; t < 16; ++t) {
#pragma unroll
    for (int r = 0; r < 8; ++r) {
      float e = __expf(sc[t][r] - rowmax[r]);
      sc[t][r] = e;
      rowsum[r] += e;
    }
  }
#pragma unroll
  for (int r = 0; r < 8; ++r) rowsum[r] = half_reduce_sum(rowsum[r]);
  if (nlo == 0) {
#pragma unroll
    for (int r = 0; r < 8; ++r) red[r + mh][wave] = rowsum[r];
  }
  __syncthreads();
#pragma unroll
  for (int r = 0; r < 8; ++r) {
    float s_ = 0.f;
#pragma unroll
    for (int w = 0; w < 8; ++w) s_ += red[r + mh][w];
    rowsum[r] = 1.0f / s_;
  }

  // phase 4: single write of normalized A
#pragma unroll
  for (int t = 0; t < 16; ++t) {
    int s = (wave * 16 + t) * 16 + nlo;
#pragma unroll
    for (int r = 0; r < 8; ++r) {
      int l = l0 + r + mh;
      A[((size_t)bh * L_ + l) * L_ + s] = sc[t][r] * rowsum[r];
    }
  }
}

// ---------------- Kernel 4: V = A@v (per head) then out = V@Wo + bo ------------
__global__ void out_kernel(const float* __restrict__ A, const __bf16* __restrict__ vT,
                           const __bf16* __restrict__ woT, const float* __restrict__ bo,
                           float* __restrict__ out) {
  __shared__ __bf16 vlds[16][D_];
  const int lane = threadIdx.x & 31;
  const int wave = threadIdx.x >> 5;   // = head in phase 1
  const int b  = blockIdx.y;
  const int l0 = blockIdx.x * 16;
  const int mh = (lane & 16) ? 8 : 0;
  const int nlo = lane & 15;

  const float*  Ah = A  + (((size_t)(b * H_ + wave) * L_) + l0) * L_;
  const __bf16* vh = vT + (size_t)(b * H_ + wave) * DH_ * L_;

  v8f c0 = {}, c1 = {}, c2 = {}, c3 = {};
  for (int ks = 0; ks < 64; ++ks) {
    int k0 = ks * 32;
    __builtin_prefetch(Ah + k0 + 64, 0, 1);
    v16bf a  = load_a_f32(Ah + k0, L_, lane);
    v16bf b0 = load_b_bf16(vh + (size_t) 0 * L_ + k0, L_, lane);
    v16bf b1 = load_b_bf16(vh + (size_t)16 * L_ + k0, L_, lane);
    v16bf b2 = load_b_bf16(vh + (size_t)32 * L_ + k0, L_, lane);
    v16bf b3 = load_b_bf16(vh + (size_t)48 * L_ + k0, L_, lane);
    c0 = wmma_bf16(a, b0, c0);
    c1 = wmma_bf16(a, b1, c1);
    c2 = wmma_bf16(a, b2, c2);
    c3 = wmma_bf16(a, b3, c3);
  }
#pragma unroll
  for (int r = 0; r < 8; ++r) {
    int m = r + mh;
    vlds[m][wave * 64 +  0 + nlo] = (__bf16)c0[r];
    vlds[m][wave * 64 + 16 + nlo] = (__bf16)c1[r];
    vlds[m][wave * 64 + 32 + nlo] = (__bf16)c2[r];
    vlds[m][wave * 64 + 48 + nlo] = (__bf16)c3[r];
  }
  __syncthreads();

  const int n0 = wave * 64;
  v8f o0 = {}, o1 = {}, o2 = {}, o3 = {};
  for (int kk = 0; kk < 16; ++kk) {
    v16bf a  = load_a_bf16(&vlds[0][0] + kk * 32, D_, lane);
    v16bf b0 = load_b_bf16(woT + (size_t)(n0 +  0) * D_ + kk * 32, D_, lane);
    v16bf b1 = load_b_bf16(woT + (size_t)(n0 + 16) * D_ + kk * 32, D_, lane);
    v16bf b2 = load_b_bf16(woT + (size_t)(n0 + 32) * D_ + kk * 32, D_, lane);
    v16bf b3 = load_b_bf16(woT + (size_t)(n0 + 48) * D_ + kk * 32, D_, lane);
    o0 = wmma_bf16(a, b0, o0);
    o1 = wmma_bf16(a, b1, o1);
    o2 = wmma_bf16(a, b2, o2);
    o3 = wmma_bf16(a, b3, o3);
  }
#pragma unroll
  for (int r = 0; r < 8; ++r) {
    int m = r + mh;
    size_t rowoff = ((size_t)b * L_ + l0 + m) * D_;
    out[rowoff + n0 +  0 + nlo] = o0[r] + bo[n0 +  0 + nlo];
    out[rowoff + n0 + 16 + nlo] = o1[r] + bo[n0 + 16 + nlo];
    out[rowoff + n0 + 32 + nlo] = o2[r] + bo[n0 + 32 + nlo];
    out[rowoff + n0 + 48 + nlo] = o3[r] + bo[n0 + 48 + nlo];
  }
}

extern "C" void kernel_launch(void* const* d_in, const int* in_sizes, int n_in,
                              void* d_out, int out_size, void* d_ws, size_t ws_size,
                              hipStream_t stream) {
  const float* x  = (const float*)d_in[0];
  // d_in[1] = key_indices (unused by the reference computation)
  const float* Wq = (const float*)d_in[2];
  const float* Wk = (const float*)d_in[3];
  const float* Wv = (const float*)d_in[4];
  const float* We = (const float*)d_in[5];
  const float* Wo = (const float*)d_in[6];
  const float* bo = (const float*)d_in[7];

  char* ws = (char*)d_ws;
  __bf16* qb  = (__bf16*)(ws);                                     // 4 MiB [B,H,L,dh]
  __bf16* kb  = (__bf16*)(ws + ((size_t)4  << 20));                // 4 MiB [B,H,L,dh]
  __bf16* vtb = (__bf16*)(ws + ((size_t)8  << 20));                // 4 MiB [B,H,dh,L]
  __bf16* wqT = (__bf16*)(ws + ((size_t)12 << 20));                // 512 KiB
  __bf16* wkT = (__bf16*)(ws + ((size_t)12 << 20) + (512u << 10));
  __bf16* wvT = (__bf16*)(ws + ((size_t)13 << 20));
  __bf16* woT = (__bf16*)(ws + ((size_t)13 << 20) + (512u << 10));
  __bf16* weT = (__bf16*)(ws + ((size_t)14 << 20));                // 256 KiB [f][d]
  __bf16* xb  = (__bf16*)(ws + ((size_t)14 << 20) + (512u << 10)); // 4 MiB

  float* out = (float*)d_out;                                      // B*L*D
  float* A   = (float*)d_out + (size_t)B_ * L_ * D_;               // B*H*L*L

  prep_kernel<<<2048, 256, 0, stream>>>(x, Wq, Wk, Wv, Wo, We,
                                        xb, wqT, wkT, wvT, woT, weT);
  qkv_kernel<<<dim3(256, 12), 256, 0, stream>>>(xb, wqT, wkT, wvT, qb, kb, vtb);
  scores_softmax_kernel<<<dim3(128, 16), 256, 0, stream>>>(qb, kb, weT, A);
  out_kernel<<<dim3(128, B_), 256, 0, stream>>>(A, vtb, woT, bo, out);
}
